// SectorGNN_89919435309475
// MI455X (gfx1250) — matrix operand order repaired and checked by reading, not compiled, not run on hardware
//
#include <hip/hip_runtime.h>
#include <hip/hip_bf16.h>
#include <hip/hip_fp16.h>

typedef __attribute__((ext_vector_type(16))) _Float16 v16h;
typedef __attribute__((ext_vector_type(8)))  _Float16 v8h;
typedef __attribute__((ext_vector_type(8)))  float    v8f;

#define NNODES 100000
#define FIN    128
#define HIDDIM 64
#define NL     4
#define NH     4
#define HDIM   16

// ---------- helpers ----------
__device__ __forceinline__ float gelu_exact(float x) {
    return 0.5f * x * (1.0f + erff(x * 0.70710678118654752f));
}
// order-preserving float<->uint mapping for atomic max on floats (incl. negatives)
__device__ __forceinline__ unsigned f2ord(float f) {
    unsigned u = __float_as_uint(f);
    return (u & 0x80000000u) ? ~u : (u | 0x80000000u);
}
__device__ __forceinline__ float ord2f(unsigned u) {
    return (u & 0x80000000u) ? __uint_as_float(u ^ 0x80000000u) : __uint_as_float(~u);
}
__device__ __forceinline__ float wave_sum(float v) {
    #pragma unroll
    for (int off = 16; off > 0; off >>= 1) v += __shfl_xor(v, off, 32);
    return v;
}

union V16U { v16h v; v8h h[2]; };

// ---------- weight prep: convert to f16, TRANSPOSED layout BT[ncols][K] ----------
// wf16 layout (half-element offsets):
//   [0)      WinT  64x128 = 8192    WinT[c][k]    = Win[k][c]
//   [8192)   WcT   L x 64x64        WcT[l][c][f]  = Wg[l][c/16][f][c%16]
//   [24576)  WoutT L x 64x64        WoutT[l][c][k]= Wout[l][k][c]
//   [40960)  We1T  64x64            We1T[c][k]    = We1[k][c]
//   [45056)  We2T  64x64            We2T[c][k]    = We2[k][c]
//   [49152)  WrpT  32x64 = 2048     WrpT[c][k]    = Wrp[k][c]
__global__ void prep_weights_kernel(const float* Win, const float* Wg, const float* Wout,
                                    const float* We1, const float* We2, const float* Wrp,
                                    _Float16* wf) {
    int tid = blockIdx.x * blockDim.x + threadIdx.x;
    const int N_WIN = HIDDIM * FIN, N_WC = NL * HIDDIM * HIDDIM, N_WOUT = N_WC;
    const int N_WE1 = HIDDIM * HIDDIM, N_WE2 = N_WE1, N_WRP = (HIDDIM / 2) * HIDDIM;
    const int total = N_WIN + N_WC + N_WOUT + N_WE1 + N_WE2 + N_WRP;
    if (tid >= total) return;
    float v; int o = tid;
    if (o < N_WIN) {
        int c = o / FIN, k = o % FIN;
        v = Win[k * HIDDIM + c];
    } else {
        o -= N_WIN;
        if (o < N_WC) {
            int l = o / (HIDDIM * HIDDIM), r = o % (HIDDIM * HIDDIM);
            int c = r / HIDDIM, f = r % HIDDIM;
            int h = c / HDIM, d = c % HDIM;
            v = Wg[(((l * NH + h) * HIDDIM) + f) * HDIM + d];
        } else {
            o -= N_WC;
            if (o < N_WOUT) {
                int l = o / (HIDDIM * HIDDIM), r = o % (HIDDIM * HIDDIM);
                int c = r / HIDDIM, k = r % HIDDIM;
                v = Wout[(l * HIDDIM + k) * HIDDIM + c];
            } else { o -= N_WOUT;
                if (o < N_WE1) { int c = o / HIDDIM, k = o % HIDDIM; v = We1[k * HIDDIM + c]; }
                else { o -= N_WE1;
                    if (o < N_WE2) { int c = o / HIDDIM, k = o % HIDDIM; v = We2[k * HIDDIM + c]; }
                    else { o -= N_WE2;
                        int c = o / HIDDIM, k = o % HIDDIM; v = Wrp[k * (HIDDIM / 2) + c];
                    }
                }
            }
        }
    }
    wf[tid] = (_Float16)v;
}

// ---------- input LayerNorm -> f16 (one wave per 128-wide row) ----------
__global__ void ln_input_kernel(const float* __restrict__ x, const float* __restrict__ g,
                                const float* __restrict__ b, _Float16* __restrict__ out) {
    int wave = (blockIdx.x * blockDim.x + threadIdx.x) >> 5;
    int lane = threadIdx.x & 31;
    if (wave >= NNODES) return;
    const float* row = x + (size_t)wave * FIN;
    float v[4]; float s = 0.f;
    #pragma unroll
    for (int j = 0; j < 4; j++) { v[j] = row[lane + 32 * j]; s += v[j]; }
    s = wave_sum(s);
    float mean = s * (1.0f / FIN), q = 0.f;
    #pragma unroll
    for (int j = 0; j < 4; j++) { float d = v[j] - mean; q += d * d; }
    q = wave_sum(q);
    float rstd = rsqrtf(q * (1.0f / FIN) + 1e-5f);
    _Float16* orow = out + (size_t)wave * FIN;
    #pragma unroll
    for (int j = 0; j < 4; j++) {
        int c = lane + 32 * j;
        orow[c] = (_Float16)((v[j] - mean) * rstd * g[c] + b[c]);
    }
}

// ---------- WMMA GEMM: one wave computes a full 16-row x NCOLS stripe ----------
// A: nrows x K row-major f16.  BT: NCOLS x K row-major f16 (i.e. B transposed).
// act: 0=none, 1=gelu, 2=tanh.  outF/outH nullable.
template <int K, int NCOLS>
__global__ void gemm_wmma_kernel(const _Float16* __restrict__ A, const _Float16* __restrict__ BT,
                                 const float* __restrict__ bias,
                                 float* __restrict__ outF, _Float16* __restrict__ outH,
                                 int nrows, int act) {
    constexpr int CT = NCOLS >> 4;
    int wid  = (blockIdx.x * blockDim.x + threadIdx.x) >> 5;
    int lane = threadIdx.x & 31;
    if (wid >= (nrows >> 4)) return;
    int r0  = wid << 4;
    int mn  = lane & 15;          // A row / B col within tile
    int hi  = lane >> 4;          // lane half
    int kbA = hi * 8;             // A: K groups 0-7/16-23 vs 8-15/24-31
    int kbB = hi * 8;             // B: K groups 0-7/16-23 vs 8-15/24-31

    v8f acc[CT];
    #pragma unroll
    for (int ct = 0; ct < CT; ct++) acc[ct] = (v8f){};

    const _Float16* Arow = A + (size_t)(r0 + mn) * K;
    #pragma unroll
    for (int kk = 0; kk < K; kk += 32) {
        V16U a;
        a.h[0] = *(const v8h*)(Arow + kk + kbA);
        a.h[1] = *(const v8h*)(Arow + kk + 16 + kbA);
        #pragma unroll
        for (int ct = 0; ct < CT; ct++) {
            const _Float16* Bcol = BT + (size_t)(ct * 16 + mn) * K + kk + kbB;
            V16U b;
            b.h[0] = *(const v8h*)(Bcol);
            b.h[1] = *(const v8h*)(Bcol + 16);
            acc[ct] = __builtin_amdgcn_wmma_f32_16x16x32_f16(false, a.v, false, b.v,
                                                             (short)0, acc[ct], false, false);
        }
    }
    #pragma unroll
    for (int ct = 0; ct < CT; ct++) {
        int col = ct * 16 + mn;
        float bv = bias ? bias[col] : 0.0f;
        #pragma unroll
        for (int g8 = 0; g8 < 8; g8++) {
            int row = r0 + g8 + hi * 8;
            float v = acc[ct][g8] + bv;
            if (act == 1) v = gelu_exact(v);
            else if (act == 2) v = tanhf(v);
            size_t idx = (size_t)row * NCOLS + col;
            if (outF) outF[idx] = v;
            if (outH) outH[idx] = (_Float16)v;
        }
    }
}

// ---------- per-node attention partials + per-layer state init ----------
__global__ void attn_prep_kernel(const float* __restrict__ hh, const float* __restrict__ attn,
                                 float* e_src, float* e_dst, float* asum,
                                 unsigned* segmax, float* agg) {
    int n = blockIdx.x * blockDim.x + threadIdx.x;
    if (n >= NNODES) return;
    const float* row = hh + (size_t)n * HIDDIM;
    #pragma unroll
    for (int h = 0; h < NH; h++) {
        float es = 0.f, ed = 0.f;
        #pragma unroll
        for (int d = 0; d < HDIM; d++) {
            float v = row[h * HDIM + d];
            es += v * attn[h * 2 * HDIM + d];
            ed += v * attn[h * 2 * HDIM + HDIM + d];
        }
        e_src[n * NH + h]  = es;
        e_dst[n * NH + h]  = ed;
        asum[n * NH + h]   = 0.f;
        segmax[n * NH + h] = f2ord(-1e30f);
    }
    float* arow = agg + (size_t)n * HIDDIM;
    #pragma unroll
    for (int c = 0; c < HIDDIM; c++) arow[c] = 0.f;
}

// ---------- edge pass 1: segment max of leaky-relu scores ----------
__global__ void edge_max_kernel(const int* __restrict__ src, const int* __restrict__ dst,
                                const unsigned char* __restrict__ mask,
                                const float* __restrict__ e_src, const float* __restrict__ e_dst,
                                unsigned* segmax, int E) {
    int e = blockIdx.x * blockDim.x + threadIdx.x;
    if (e >= E) return;
    int s = __builtin_nontemporal_load(&src[e]);
    int d = __builtin_nontemporal_load(&dst[e]);
    if (!mask[s]) return;
    #pragma unroll
    for (int h = 0; h < NH; h++) {
        float sc = e_src[s * NH + h] + e_dst[d * NH + h];
        sc = sc >= 0.f ? sc : 0.2f * sc;
        atomicMax(&segmax[d * NH + h], f2ord(sc));
    }
}

// ---------- edge pass 2: alpha = exp(min(e-max,20)); scatter-add alpha and alpha*hs ----------
__global__ void edge_accum_kernel(const int* __restrict__ src, const int* __restrict__ dst,
                                  const unsigned char* __restrict__ mask,
                                  const float* __restrict__ e_src, const float* __restrict__ e_dst,
                                  const unsigned* __restrict__ segmax,
                                  float* asum, float* agg, const float* __restrict__ hh, int E) {
    int e = blockIdx.x * blockDim.x + threadIdx.x;
    if (e >= E) return;
    int s = __builtin_nontemporal_load(&src[e]);
    int d = __builtin_nontemporal_load(&dst[e]);
    if (!mask[s]) return;
    const float* hrow = hh + (size_t)s * HIDDIM;
    #pragma unroll
    for (int h = 0; h < NH; h++) {
        float sc = e_src[s * NH + h] + e_dst[d * NH + h];
        sc = sc >= 0.f ? sc : 0.2f * sc;
        float m = ord2f(segmax[d * NH + h]);
        if (m < -1e29f) m = 0.f;  // reference: non-finite segmax -> 0
        float alpha = expf(fminf(sc - m, 20.0f));
        atomicAdd(&asum[d * NH + h], alpha);
        #pragma unroll
        for (int dd = 0; dd < HDIM; dd++) {
            atomicAdd(&agg[(size_t)d * HIDDIM + h * HDIM + dd], alpha * hrow[h * HDIM + dd]);
        }
    }
}

// ---------- normalize aggregate by alpha-sum, convert to f16 for next GEMM ----------
__global__ void agg_norm_kernel(const float* __restrict__ agg, const float* __restrict__ asum,
                                _Float16* __restrict__ out) {
    size_t i = (size_t)blockIdx.x * blockDim.x + threadIdx.x;
    if (i >= (size_t)NNODES * HIDDIM) return;
    int n = (int)(i >> 6);
    int c = (int)(i & 63);
    int h = c >> 4;
    out[i] = (_Float16)(agg[i] / (asum[n * NH + h] + 1e-8f));
}

// ---------- residual + LayerNorm (one wave per 64-wide row), dual f32/f16 write ----------
__global__ void res_ln_kernel(float* __restrict__ hf, const float* __restrict__ t,
                              const float* __restrict__ g, const float* __restrict__ b,
                              _Float16* __restrict__ h16) {
    int wave = (blockIdx.x * blockDim.x + threadIdx.x) >> 5;
    int lane = threadIdx.x & 31;
    if (wave >= NNODES) return;
    float* row = hf + (size_t)wave * HIDDIM;
    const float* trow = t + (size_t)wave * HIDDIM;
    float v0 = row[lane] + trow[lane];
    float v1 = row[lane + 32] + trow[lane + 32];
    float s = wave_sum(v0 + v1);
    float mean = s * (1.0f / HIDDIM);
    float d0 = v0 - mean, d1 = v1 - mean;
    float q = wave_sum(d0 * d0 + d1 * d1);
    float rstd = rsqrtf(q * (1.0f / HIDDIM) + 1e-5f);
    float y0 = d0 * rstd * g[lane] + b[lane];
    float y1 = d1 * rstd * g[lane + 32] + b[lane + 32];
    row[lane] = y0; row[lane + 32] = y1;
    _Float16* hrow = h16 + (size_t)wave * HIDDIM;
    hrow[lane] = (_Float16)y0; hrow[lane + 32] = (_Float16)y1;
}

// ---------- pred head: proj@Wro + bro + h@Wrs ----------
__global__ void pred_kernel(const _Float16* __restrict__ proj, const float* __restrict__ hf,
                            const float* __restrict__ Wro, const float* __restrict__ bro,
                            const float* __restrict__ Wrs, float* __restrict__ out) {
    int n = blockIdx.x * blockDim.x + threadIdx.x;
    if (n >= NNODES) return;
    const _Float16* prow = proj + (size_t)n * 32;
    const float* hrow = hf + (size_t)n * 64;
    float acc = bro[0];
    #pragma unroll
    for (int i = 0; i < 32; i++) acc += (float)prow[i] * Wro[i];
    #pragma unroll
    for (int i = 0; i < 64; i++) acc += hrow[i] * Wrs[i];
    out[n] = acc;
}

extern "C" void kernel_launch(void* const* d_in, const int* in_sizes, int n_in,
                              void* d_out, int out_size, void* d_ws, size_t ws_size,
                              hipStream_t stream) {
    (void)n_in; (void)out_size; (void)ws_size;
    const float* x    = (const float*)d_in[0];
    const int*   ei   = (const int*)d_in[1];
    const unsigned char* mask = (const unsigned char*)d_in[2];
    const float* lng  = (const float*)d_in[3];
    const float* lnb  = (const float*)d_in[4];
    const float* Win  = (const float*)d_in[5];
    const float* bin  = (const float*)d_in[6];
    const float* Wg   = (const float*)d_in[7];
    const float* attn = (const float*)d_in[8];
    const float* Wout = (const float*)d_in[9];
    const float* ng   = (const float*)d_in[10];
    const float* nb   = (const float*)d_in[11];
    const float* We1  = (const float*)d_in[12];
    const float* be1  = (const float*)d_in[13];
    const float* We2  = (const float*)d_in[14];
    const float* be2  = (const float*)d_in[15];
    const float* Wrp  = (const float*)d_in[16];
    const float* brp  = (const float*)d_in[17];
    const float* Wro  = (const float*)d_in[18];
    const float* bro  = (const float*)d_in[19];
    const float* Wrs  = (const float*)d_in[20];
    const int E = in_sizes[1] / 2;

    // workspace carve-up (256B aligned regions)
    char* ws = (char*)d_ws;
    size_t off = 0;
    auto carve = [&](size_t bytes) -> char* {
        char* p = ws + off;
        off += (bytes + 255) & ~(size_t)255;
        return p;
    };
    float*    h_f32   = (float*)carve((size_t)NNODES * 64 * 4);
    float*    hh_f32  = (float*)carve((size_t)NNODES * 64 * 4);   // also reused for t = agg@Wout
    float*    agg_f32 = (float*)carve((size_t)NNODES * 64 * 4);
    float*    e_src   = (float*)carve((size_t)NNODES * 4 * 4);
    float*    e_dst   = (float*)carve((size_t)NNODES * 4 * 4);
    float*    asum    = (float*)carve((size_t)NNODES * 4 * 4);
    unsigned* segmax  = (unsigned*)carve((size_t)NNODES * 4 * 4);
    _Float16* lnx     = (_Float16*)carve((size_t)NNODES * 128 * 2);
    _Float16* h16     = (_Float16*)carve((size_t)NNODES * 64 * 2);
    _Float16* agg16   = (_Float16*)carve((size_t)NNODES * 64 * 2); // also reused for t1 = gelu(h@We1)
    _Float16* proj16  = (_Float16*)carve((size_t)NNODES * 32 * 2);
    _Float16* wf16    = (_Float16*)carve((size_t)51200 * 2);

    const _Float16* WF_WIN  = wf16;            // 64x128 (transposed)
    const _Float16* WF_WC   = wf16 + 8192;     // L x 64x64 (transposed)
    const _Float16* WF_WOUT = wf16 + 24576;    // L x 64x64 (transposed)
    const _Float16* WF_WE1  = wf16 + 40960;    // 64x64 (transposed)
    const _Float16* WF_WE2  = wf16 + 45056;    // 64x64 (transposed)
    const _Float16* WF_WRP  = wf16 + 49152;    // 32x64 (transposed)

    float* pred_out = (float*)d_out;
    float* emb_out  = (float*)d_out + NNODES;

    const int WROWS = NNODES / 16;             // 6250 waves: one 16-row stripe each
    const int BGEMM = (WROWS + 7) / 8;
    const int BN    = (NNODES + 255) / 256;
    const int BE    = (E + 255) / 256;
    const int BROW  = NNODES / 8;              // wave-per-row kernels: 100000 waves

    prep_weights_kernel<<<(51200 + 255) / 256, 256, 0, stream>>>(Win, Wg, Wout, We1, We2, Wrp, wf16);
    ln_input_kernel<<<BROW, 256, 0, stream>>>(x, lng, lnb, lnx);
    // h = gelu(LN(x) @ W_in + b_in)
    gemm_wmma_kernel<128, 64><<<BGEMM, 256, 0, stream>>>(lnx, WF_WIN, bin, h_f32, h16, NNODES, 1);

    for (int l = 0; l < NL; l++) {
        gemm_wmma_kernel<64, 64><<<BGEMM, 256, 0, stream>>>(h16, WF_WC + l * 4096, nullptr,
                                                            hh_f32, nullptr, NNODES, 0);
        attn_prep_kernel<<<BN, 256, 0, stream>>>(hh_f32, attn + l * (NH * 2 * HDIM),
                                                 e_src, e_dst, asum, segmax, agg_f32);
        edge_max_kernel<<<BE, 256, 0, stream>>>(ei, ei + E, mask, e_src, e_dst, segmax, E);
        edge_accum_kernel<<<BE, 256, 0, stream>>>(ei, ei + E, mask, e_src, e_dst, segmax,
                                                  asum, agg_f32, hh_f32, E);
        agg_norm_kernel<<<(NNODES * 64 + 255) / 256, 256, 0, stream>>>(agg_f32, asum, agg16);
        gemm_wmma_kernel<64, 64><<<BGEMM, 256, 0, stream>>>(agg16, WF_WOUT + l * 4096, nullptr,
                                                            hh_f32, nullptr, NNODES, 0);
        res_ln_kernel<<<BROW, 256, 0, stream>>>(h_f32, hh_f32, ng + l * 64, nb + l * 64, h16);
    }

    // emb = tanh(gelu(h@We1+be1)@We2+be2)
    gemm_wmma_kernel<64, 64><<<BGEMM, 256, 0, stream>>>(h16, WF_WE1, be1, nullptr, agg16, NNODES, 1);
    gemm_wmma_kernel<64, 64><<<BGEMM, 256, 0, stream>>>(agg16, WF_WE2, be2, emb_out, nullptr, NNODES, 2);
    // pred = gelu(h@Wrp+brp)@Wro + bro + h@Wrs
    gemm_wmma_kernel<64, 32><<<BGEMM, 256, 0, stream>>>(h16, WF_WRP, brp, nullptr, proj16, NNODES, 1);
    pred_kernel<<<BN, 256, 0, stream>>>(proj16, h_f32, Wro, bro, Wrs, pred_out);
}
